// MutualInformation_62792421868328
// MI455X (gfx1250) — matrix operand order repaired and checked by reading, not compiled, not run on hardware
//
#include <hip/hip_runtime.h>

// ---------------------------------------------------------------------------
// KDE mutual information on gfx1250 (MI455X), separable-kernel GEMM via WMMA.
//
//   p_xy = (coef/N) * Kx (20 x K) * Ky^T (K x 20),  K = N samples
//   Kx[i,n] = exp(-(i/20 - x_n)^2 / (2 var)) = exp2(-(i/20 - x_n)^2 * S)
//
// 80M MACs -> v_wmma_f32_16x16x32_f16 (f32 accum), 12.8M exps -> v_exp_f32.
// 1.6 MB of input: fully L2 resident; kernel is VALU/XDL bound.
// K-loop is a counted loop with an SGPR trip count (readfirstlane), so all
// control flow is scalar and EXEC is all-ones at every WMMA.
// Tile-1 rows >= 20 are zeroed via a loop-invariant offset that drives the
// exp argument to -inf (exp2 -> 0), not per-element cndmasks.
// ---------------------------------------------------------------------------

typedef __attribute__((ext_vector_type(16))) _Float16 v16h;
typedef __attribute__((ext_vector_type(8)))  float    v8f;
typedef __attribute__((ext_vector_type(4)))  float    v4f;

#define NG   20
#define GG   (NG * NG)

// S = log2(e) / (2 * 0.04^2)
#define KDE_S    450.84219965f
// coef = 1/sqrt(2*pi*var)
#define KDE_COEF 9.9735570100f
#define LN2      0.69314718056f

__global__ void zero_ws_kernel(float* ws) {
    int t = blockIdx.x * blockDim.x + threadIdx.x;
    if (t < GG) ws[t] = 0.0f;
}

// Build A0/A1 (x-kernel rows) and B0/B1 (y-kernel cols) fragments from 32
// samples and accumulate the four 16x16 tiles.  bias1 = 0.8 for valid tile-1
// rows, 1e19 for padding rows (squares to ~1e38, *(-S) -> -inf, exp2 -> 0).
__device__ __forceinline__ void frag_mma(
    const float xv[16], const float yv[16], float g0, float bias1,
    v8f& a00, v8f& a01, v8f& a10, v8f& a11) {

    v16h A0, A1, B0, B1;
#pragma unroll
    for (int e = 0; e < 16; ++e) {
        float dx0 = g0 - xv[e];
        float dx1 = dx0 + bias1;
        float dy0 = g0 - yv[e];
        float dy1 = dy0 + bias1;
        A0[e] = (_Float16)__builtin_amdgcn_exp2f(-(dx0 * dx0) * KDE_S);
        A1[e] = (_Float16)__builtin_amdgcn_exp2f(-(dx1 * dx1) * KDE_S);
        B0[e] = (_Float16)__builtin_amdgcn_exp2f(-(dy0 * dy0) * KDE_S);
        B1[e] = (_Float16)__builtin_amdgcn_exp2f(-(dy1 * dy1) * KDE_S);
    }
    a00 = __builtin_amdgcn_wmma_f32_16x16x32_f16(false, A0, false, B0,
                                                 (short)0, a00, false, false);
    a01 = __builtin_amdgcn_wmma_f32_16x16x32_f16(false, A0, false, B1,
                                                 (short)0, a01, false, false);
    a10 = __builtin_amdgcn_wmma_f32_16x16x32_f16(false, A1, false, B0,
                                                 (short)0, a10, false, false);
    a11 = __builtin_amdgcn_wmma_f32_16x16x32_f16(false, A1, false, B1,
                                                 (short)0, a11, false, false);
}

__global__ __launch_bounds__(256) void kde_wmma_kernel(
    const float* __restrict__ bx, const float* __restrict__ by,
    float* __restrict__ ws, int Ntot) {

    __shared__ float s_p[GG];
    for (int t = threadIdx.x; t < GG; t += blockDim.x) s_p[t] = 0.0f;
    __syncthreads();

    const int lane = threadIdx.x & 31;
    const int grp  = lane >> 4;            // half-wave group (ISA fragment layout)
    const int m    = lane & 15;            // A row / B column within a 16x16 tile

    const int wid    = __builtin_amdgcn_readfirstlane(
        (int)blockIdx.x * (int)(blockDim.x >> 5) + (int)(threadIdx.x >> 5));
    const int nWaves = __builtin_amdgcn_readfirstlane(
        (int)gridDim.x * (int)(blockDim.x >> 5));

    const int fullChunks = Ntot >> 5;      // complete 32-sample chunks
    int myIters = (fullChunks > wid) ? (fullChunks - wid + nWaves - 1) / nWaves : 0;
    myIters = __builtin_amdgcn_readfirstlane(myIters);   // SGPR trip count

    const float g0    = (float)m * (1.0f / NG);
    const float bias1 = ((m + 16) < NG) ? 0.8f : 1.0e19f;

    // Running fragment pointers (A and B: K = grp*8 + {0..7,16..23}).
    const float* pA = bx + grp * 8  + (size_t)wid * 32;
    const float* pB = by + grp * 8  + (size_t)wid * 32;
    const int strideF = nWaves * 32;

    v8f acc00 = {}, acc01 = {}, acc10 = {}, acc11 = {};

    for (int it = 0; it < myIters; ++it) {
        // Speculative prefetch of this wave's next chunk (dropped if OOB).
        __builtin_prefetch(pA + strideF, 0, 1);
        __builtin_prefetch(pB + strideF, 0, 1);

        v4f t0 = *(const v4f*)(pA);
        v4f t1 = *(const v4f*)(pA + 4);
        v4f t2 = *(const v4f*)(pA + 16);
        v4f t3 = *(const v4f*)(pA + 20);
        v4f u0 = *(const v4f*)(pB);
        v4f u1 = *(const v4f*)(pB + 4);
        v4f u2 = *(const v4f*)(pB + 16);
        v4f u3 = *(const v4f*)(pB + 20);
        pA += strideF;
        pB += strideF;

        float xv[16], yv[16];
#pragma unroll
        for (int j = 0; j < 4; ++j) {
            xv[j] = t0[j]; xv[4 + j] = t1[j]; xv[8 + j] = t2[j]; xv[12 + j] = t3[j];
            yv[j] = u0[j]; yv[4 + j] = u1[j]; yv[8 + j] = u2[j]; yv[12 + j] = u3[j];
        }
        frag_mma(xv, yv, g0, bias1, acc00, acc01, acc10, acc11);
    }

    // Tail chunk (Ntot % 32 != 0), handled once by its owning wave.
    const int rem = Ntot & 31;
    if (__builtin_amdgcn_readfirstlane(
            (rem != 0 && wid == (fullChunks % nWaves)) ? 1 : 0)) {
        const int base = fullChunks * 32;
        float xv[16], yv[16];
#pragma unroll
        for (int e = 0; e < 16; ++e) {
            int ka = base + grp * 8 + e + (e < 8 ? 0 : 8);
            int kb = base + grp * 8 + e + (e < 8 ? 0 : 8);
            // Sentinel: kernel value becomes exactly 0 (exp2(-inf)).
            xv[e] = (ka < Ntot) ? bx[ka] : 3.0e18f;
            yv[e] = (kb < Ntot) ? by[kb] : 3.0e18f;
        }
        frag_mma(xv, yv, g0, bias1, acc00, acc01, acc10, acc11);
    }

    // C/D fragment layout: VGPR r, lanes 0-15 -> M=r, lanes 16-31 -> M=r+8.
#pragma unroll
    for (int ti = 0; ti < 2; ++ti) {
#pragma unroll
        for (int tj = 0; tj < 2; ++tj) {
            const v8f& a = (ti == 0) ? ((tj == 0) ? acc00 : acc01)
                                     : ((tj == 0) ? acc10 : acc11);
            int col = m + 16 * tj;
            if (col < NG) {
#pragma unroll
                for (int r = 0; r < 8; ++r) {
                    int row = r + 8 * grp + 16 * ti;
                    if (row < NG) atomicAdd(&s_p[row * NG + col], a[r]);  // ds_add_f32
                }
            }
        }
    }
    __syncthreads();
    for (int t = threadIdx.x; t < GG; t += blockDim.x)
        atomicAdd(ws + t, s_p[t]);  // global_atomic_add_f32, one per cell per block
}

__global__ __launch_bounds__(512) void mi_finalize_kernel(
    const float* __restrict__ ws, float* __restrict__ out, int Ntot) {

    __shared__ float sp[GG];
    __shared__ float spx[NG];
    __shared__ float spy[NG];
    __shared__ float s_acc;

    const int t = threadIdx.x;
    const float scale = KDE_COEF / (float)Ntot;
    for (int i = t; i < GG; i += blockDim.x) sp[i] = ws[i] * scale;
    if (t == 0) s_acc = 0.0f;
    __syncthreads();

    if (t < NG) {                        // row marginals
        float s = 0.0f;
        for (int j = 0; j < NG; ++j) s += sp[t * NG + j];
        spx[t] = s;
    } else if (t >= 32 && t < 32 + NG) { // column marginals
        int i2 = t - 32;
        float s = 0.0f;
        for (int j = 0; j < NG; ++j) s += sp[j * NG + i2];
        spy[i2] = s;
    }
    __syncthreads();

    float part = 0.0f;
    for (int i = t; i < GG; i += blockDim.x) {
        float p = sp[i];
        if (p > 0.0f) {
            float r = p / (spx[i / NG] * spy[i % NG]);
            part += p * (__builtin_amdgcn_logf(r) * LN2);  // ln = log2 * ln2
        }
    }
    atomicAdd(&s_acc, part);
    __syncthreads();
    if (t == 0) out[0] = s_acc;
}

extern "C" void kernel_launch(void* const* d_in, const int* in_sizes, int n_in,
                              void* d_out, int out_size, void* d_ws, size_t ws_size,
                              hipStream_t stream) {
    const float* b1 = (const float*)d_in[0];
    const float* b2 = (const float*)d_in[1];
    float* out = (float*)d_out;
    float* ws  = (float*)d_ws;
    const int Ntot    = in_sizes[0];
    const int nChunks = (Ntot + 31) / 32;

    // Workspace accumulators must start at zero every launch (graph-safe).
    zero_ws_kernel<<<1, 512, 0, stream>>>(ws);

    int blocks = 192;                               // 192 blocks * 8 waves = 1536 waves
    int maxBlocks = (nChunks + 7) / 8;              // >=1 chunk per wave
    if (blocks > maxBlocks) blocks = maxBlocks;
    if (blocks < 1) blocks = 1;
    kde_wmma_kernel<<<blocks, 256, 0, stream>>>(b1, b2, ws, Ntot);

    mi_finalize_kernel<<<1, 512, 0, stream>>>(ws, out, Ntot);
}
